// GatedGraphRecurrentLayer_21440476742385
// MI455X (gfx1250) — matrix operand order repaired and checked by reading, not compiled, not run on hardware
//
#include <hip/hip_runtime.h>
#include <stdint.h>

#define NN 100000
#define HH 256
#define H3 768
#define EE 1600000
#define MTILES (NN / 16)   // 6250, exact
#define LPAD 264           // LDS row pitch in u16 (528B): bank-conflict-free b128 reads

typedef unsigned short u16t;
typedef __attribute__((ext_vector_type(16))) __bf16 v16bf;
typedef __attribute__((ext_vector_type(8)))  float  v8f;

union Frag16 { v16bf v; uint4 q[2]; };
union Acc8   { v8f  v; float f[8]; };

__device__ __forceinline__ u16t f2bf(float f) {
    union { float f; uint32_t u; } c; c.f = f;
    uint32_t b = c.u;
    b += 0x7fffu + ((b >> 16) & 1u);      // round-to-nearest-even
    return (u16t)(b >> 16);
}
__device__ __forceinline__ float sigmoidf(float x) {
    return 1.0f / (1.0f + __expf(-x));
}

// Async copy 16 bytes: global -> LDS (per-lane addresses, tracked by ASYNCcnt)
__device__ __forceinline__ void async_g2l_b128(uint32_t lds_off, const void* gaddr) {
    asm volatile("global_load_async_to_lds_b128 %0, %1, off"
                 :: "v"(lds_off), "v"((unsigned long long)(uintptr_t)gaddr)
                 : "memory");
}
__device__ __forceinline__ void wait_async0() {
    asm volatile("s_wait_asynccnt 0x0" ::: "memory");
}

// ---------------- one-time setup kernels ----------------

__global__ void k_set_ones(float* p, int n) {
    int i = blockIdx.x * blockDim.x + threadIdx.x;
    if (i < n) p[i] = 1.0f;
}

__global__ void k_count_deg(const int* __restrict__ e0, const int* __restrict__ e1,
                            const int* __restrict__ e2, float* __restrict__ deg) {
    int i = blockIdx.x * blockDim.x + threadIdx.x;
    if (i >= 3 * EE) return;
    int t = i / EE, e = i - t * EE;
    const int* ed = (t == 0) ? e0 : (t == 1) ? e1 : e2;
    int d = ed[EE + e];                     // row 1 = dst
    atomicAdd(&deg[t * NN + d], 1.0f);
}

__global__ void k_rsqrt_ip(float* p, int n) {
    int i = blockIdx.x * blockDim.x + threadIdx.x;
    if (i < n) p[i] = rsqrtf(p[i]);        // deg >= 1 due to self loops
}

// Pack a [K=256, Ncols] matrix (or its transpose source [Ncols,256]) into
// WMMA-B fragment order: out[((ntile*8 + k0s)*32 + lane)*16 + i]
//   n = ntile*16 + (lane&15);  k = k0s*32 + (i<8 ? 8*(lane>>4) + i : 16 + 8*(lane>>4) + (i-8))
__global__ void k_pack_b(const float* __restrict__ W, u16t* __restrict__ out,
                         int ntiles, int transpose) {
    int idx = blockIdx.x * blockDim.x + threadIdx.x;
    int total = ntiles * 8 * 32 * 16;
    if (idx >= total) return;
    int i    = idx & 15;
    int lane = (idx >> 4) & 31;
    int k0s  = (idx >> 9) & 7;
    int nt   = idx >> 12;
    int n = nt * 16 + (lane & 15);
    int k = k0s * 32 + (i < 8 ? 8 * (lane >> 4) + i : 16 + 8 * (lane >> 4) + (i - 8));
    float v = transpose ? W[(size_t)n * 256 + k]
                        : W[(size_t)k * (ntiles * 16) + n];
    out[idx] = f2bf(v);
}

__global__ void k_cvt_bf16(const float* __restrict__ in, u16t* __restrict__ out, long n) {
    long i = (long)blockIdx.x * blockDim.x + threadIdx.x;
    if (i < n) out[i] = f2bf(in[i]);
}

// ---------------- per-layer kernels ----------------

// xw[t] = hbf @ W_t  (bf16 x bf16 -> f32).
// Block = 8 waves, one (mtile, t); A tile async-staged to LDS once,
// each wave computes two 16x16 N-tiles (A fragment reused for 2 WMMAs).
__global__ void k_gemm_xw(const u16t* __restrict__ hbf, const u16t* __restrict__ Wp,
                          float* __restrict__ xw) {
    __shared__ u16t At[16 * LPAD];
    int mtile = blockIdx.x;
    int t     = blockIdx.y;
    int tid   = threadIdx.x;
    int wave  = tid >> 5;
    int lane  = tid & 31;
    int half  = lane >> 4;

    {   // stage 16x256 bf16 A tile: thread copies one 32B chunk (2 x b128 async)
        int row = tid >> 4, chunk = tid & 15;
        const u16t* g = hbf + (size_t)(mtile * 16 + row) * HH + chunk * 16;
        uint32_t l = (uint32_t)(uintptr_t)(At + row * LPAD + chunk * 16);
        async_g2l_b128(l,      g);
        async_g2l_b128(l + 16, g + 8);
    }
    wait_async0();
    __syncthreads();

    const u16t* Wt = Wp + (size_t)t * 65536;
    const u16t* aL = At + (lane & 15) * LPAD;
    int n0 = wave, n1 = wave + 8;
    v8f acc0 = {}, acc1 = {};
#pragma unroll
    for (int k0s = 0; k0s < 8; ++k0s) {
        int k0 = k0s * 32;
        Frag16 af, b0, b1;
        af.q[0] = *(const uint4*)(aL + k0 + 8 * half);            // ds_load_b128
        af.q[1] = *(const uint4*)(aL + k0 + 16 + 8 * half);
        const u16t* p0 = Wt + (((size_t)n0 * 8 + k0s) * 32 + lane) * 16;
        const u16t* p1 = Wt + (((size_t)n1 * 8 + k0s) * 32 + lane) * 16;
        b0.q[0] = *(const uint4*)(p0); b0.q[1] = *(const uint4*)(p0 + 8);
        b1.q[0] = *(const uint4*)(p1); b1.q[1] = *(const uint4*)(p1 + 8);
        acc0 = __builtin_amdgcn_wmma_f32_16x16x32_bf16(
            false, af.v, false, b0.v, (short)0, acc0, false, false);
        acc1 = __builtin_amdgcn_wmma_f32_16x16x32_bf16(
            false, af.v, false, b1.v, (short)0, acc1, false, false);
    }
    Acc8 o0, o1; o0.v = acc0; o1.v = acc1;
    float* outb = xw + (size_t)t * NN * HH;
    int c0 = n0 * 16 + (lane & 15), c1 = n1 * 16 + (lane & 15);
#pragma unroll
    for (int r = 0; r < 8; ++r) {
        int row = mtile * 16 + 8 * half + r;
        outb[(size_t)row * HH + c0] = o0.f[r];
        outb[(size_t)row * HH + c1] = o1.f[r];
    }
}

// agg[n,c] = sum_t b_t[c] + sum_t dinv_t[n]^2 * xw_t[n,c]   (self-loop + bias)
__global__ void k_init_agg(const float* __restrict__ xw, const float* __restrict__ dinv,
                           const float* __restrict__ b0, const float* __restrict__ b1,
                           const float* __restrict__ b2, float* __restrict__ agg) {
    int n = blockIdx.x, c = threadIdx.x;
    float v = b0[c] + b1[c] + b2[c];
#pragma unroll
    for (int t = 0; t < 3; ++t) {
        float di = dinv[t * NN + n];
        v += di * di * xw[(size_t)t * NN * HH + (size_t)n * HH + c];
    }
    agg[(size_t)n * HH + c] = v;
}

// One wave per edge: agg[dst] += dinv[src]*dinv[dst] * xw[src]
__global__ void k_scatter(const int* __restrict__ e0, const int* __restrict__ e1,
                          const int* __restrict__ e2, const float* __restrict__ dinv,
                          const float* __restrict__ xw, float* __restrict__ agg) {
    int t    = blockIdx.z;
    int wave = threadIdx.x >> 5;
    int lane = threadIdx.x & 31;
    long e   = (long)blockIdx.x * 8 + wave;          // grid sized exactly EE/8
    const int* ed = (t == 0) ? e0 : (t == 1) ? e1 : e2;
    int s = ed[e];
    int d = ed[EE + e];
    float w = dinv[t * NN + s] * dinv[t * NN + d];
    const float* xr = xw + (size_t)t * NN * HH + (size_t)s * HH + lane * 8;
    float*       ar = agg + (size_t)d * HH + lane * 8;
    __builtin_prefetch(ar, 1, 0);                    // global_prefetch_b8
    float4 v0 = *(const float4*)(xr);
    float4 v1 = *(const float4*)(xr + 4);
    atomicAdd(ar + 0, w * v0.x); atomicAdd(ar + 1, w * v0.y);
    atomicAdd(ar + 2, w * v0.z); atomicAdd(ar + 3, w * v0.w);
    atomicAdd(ar + 4, w * v1.x); atomicAdd(ar + 5, w * v1.y);
    atomicAdd(ar + 6, w * v1.z); atomicAdd(ar + 7, w * v1.w);
}

// Fused GRU: gi = a@Wi^T + bi, gh = h@Wh^T + bh, gate, write h_new.
// a/h tiles async-staged to LDS; wave carries 6 f32 accumulators per tile.
__global__ void k_gru(const u16t* __restrict__ abf, const u16t* __restrict__ hbf,
                      const float* __restrict__ h_old,
                      const u16t* __restrict__ WiTp, const u16t* __restrict__ WhTp,
                      const float* __restrict__ bi, const float* __restrict__ bh,
                      float* __restrict__ h_new) {
    __shared__ u16t At[16 * LPAD];
    __shared__ u16t Ht[16 * LPAD];
    int mtile = blockIdx.x;
    int tid   = threadIdx.x;
    int wave  = tid >> 5;
    int lane  = tid & 31;
    int half  = lane >> 4;

    {   // stage both 16x256 bf16 tiles (4 x b128 async per thread)
        int row = tid >> 4, chunk = tid & 15;
        size_t goff = (size_t)(mtile * 16 + row) * HH + chunk * 16;
        uint32_t loff = (uint32_t)(row * LPAD + chunk * 16);
        uint32_t la = (uint32_t)(uintptr_t)(At + loff);
        uint32_t lh = (uint32_t)(uintptr_t)(Ht + loff);
        async_g2l_b128(la,      abf + goff);
        async_g2l_b128(la + 16, abf + goff + 8);
        async_g2l_b128(lh,      hbf + goff);
        async_g2l_b128(lh + 16, hbf + goff + 8);
    }
    wait_async0();
    __syncthreads();

    const u16t* aL = At + (lane & 15) * LPAD;
    const u16t* hL = Ht + (lane & 15) * LPAD;

    for (int pass = 0; pass < 2; ++pass) {
        int ntile = wave + 8 * pass;                 // 0..15 over H columns
        v8f acc[6] = {{}, {}, {}, {}, {}, {}};
#pragma unroll
        for (int k0s = 0; k0s < 8; ++k0s) {
            int k0 = k0s * 32;
            Frag16 aa, ah;
            aa.q[0] = *(const uint4*)(aL + k0 + 8 * half);
            aa.q[1] = *(const uint4*)(aL + k0 + 16 + 8 * half);
            ah.q[0] = *(const uint4*)(hL + k0 + 8 * half);
            ah.q[1] = *(const uint4*)(hL + k0 + 16 + 8 * half);
#pragma unroll
            for (int g = 0; g < 3; ++g) {
                int jt = g * 16 + ntile;             // tile in [0,48) over 3H cols
                Frag16 bwi, bwh;
                const u16t* pi = WiTp + (((size_t)jt * 8 + k0s) * 32 + lane) * 16;
                const u16t* ph = WhTp + (((size_t)jt * 8 + k0s) * 32 + lane) * 16;
                bwi.q[0] = *(const uint4*)(pi);  bwi.q[1] = *(const uint4*)(pi + 8);
                bwh.q[0] = *(const uint4*)(ph);  bwh.q[1] = *(const uint4*)(ph + 8);
                acc[g] = __builtin_amdgcn_wmma_f32_16x16x32_bf16(
                    false, aa.v, false, bwi.v, (short)0, acc[g], false, false);
                acc[3 + g] = __builtin_amdgcn_wmma_f32_16x16x32_bf16(
                    false, ah.v, false, bwh.v, (short)0, acc[3 + g], false, false);
            }
        }
        int c = ntile * 16 + (lane & 15);
        float bir = bi[c], biz = bi[256 + c], bin_ = bi[512 + c];
        float bhr = bh[c], bhz = bh[256 + c], bhn = bh[512 + c];
        Acc8 A[6];
#pragma unroll
        for (int q = 0; q < 6; ++q) A[q].v = acc[q];
#pragma unroll
        for (int r = 0; r < 8; ++r) {
            int row  = mtile * 16 + 8 * half + r;
            float ir = A[0].f[r] + bir, iz = A[1].f[r] + biz, in_ = A[2].f[r] + bin_;
            float hr = A[3].f[r] + bhr, hz = A[4].f[r] + bhz, hn  = A[5].f[r] + bhn;
            float rg = sigmoidf(ir + hr);
            float zg = sigmoidf(iz + hz);
            float ng = tanhf(in_ + rg * hn);
            float ho = h_old[(size_t)row * HH + c];
            h_new[(size_t)row * HH + c] = (1.0f - zg) * ng + zg * ho;
        }
    }
}

// ---------------- host side ----------------

extern "C" void kernel_launch(void* const* d_in, const int* in_sizes, int n_in,
                              void* d_out, int out_size, void* d_ws, size_t ws_size,
                              hipStream_t stream) {
    const float* x     = (const float*)d_in[0];
    const int*   e_ast = (const int*)d_in[1];
    const int*   e_cfg = (const int*)d_in[2];
    const int*   e_dfg = (const int*)d_in[3];
    const float* W_ast = (const float*)d_in[4];  const float* b_ast = (const float*)d_in[5];
    const float* W_cfg = (const float*)d_in[6];  const float* b_cfg = (const float*)d_in[7];
    const float* W_dfg = (const float*)d_in[8];  const float* b_dfg = (const float*)d_in[9];
    const float* Wi    = (const float*)d_in[10]; const float* Wh    = (const float*)d_in[11];
    const float* bi    = (const float*)d_in[12]; const float* bh    = (const float*)d_in[13];

    char* ws = (char*)d_ws;
    size_t off = 0;
    auto carve = [&](size_t bytes) -> char* {
        char* p = ws + off;
        off += (bytes + 255) & ~(size_t)255;
        return p;
    };
    float* dinv  = (float*)carve((size_t)3 * NN * 4);
    u16t*  Wp    = (u16t*) carve((size_t)3 * 65536 * 2);      // 3 x [16][8][32][16]
    u16t*  WiTp  = (u16t*) carve((size_t)48 * 4096 * 2);      // [48][8][32][16]
    u16t*  WhTp  = (u16t*) carve((size_t)48 * 4096 * 2);
    u16t*  hbf   = (u16t*) carve((size_t)NN * HH * 2);
    u16t*  abf   = (u16t*) carve((size_t)NN * HH * 2);
    float* xw    = (float*)carve((size_t)3 * NN * HH * 4);
    float* agg   = (float*)carve((size_t)NN * HH * 4);
    float* hA    = (float*)carve((size_t)NN * HH * 4);
    float* hB    = (float*)carve((size_t)NN * HH * 4);

    // --- degrees / normalization (edge structure is layer-invariant) ---
    k_set_ones<<<(3 * NN + 255) / 256, 256, 0, stream>>>(dinv, 3 * NN);
    k_count_deg<<<(3 * EE + 255) / 256, 256, 0, stream>>>(e_ast, e_cfg, e_dfg, dinv);
    k_rsqrt_ip<<<(3 * NN + 255) / 256, 256, 0, stream>>>(dinv, 3 * NN);

    // --- pack weights into WMMA-B fragment order (bf16), once ---
    int pb16 = 16 * 4096, pb48 = 48 * 4096;
    k_pack_b<<<(pb16 + 255) / 256, 256, 0, stream>>>(W_ast, Wp + 0 * 65536, 16, 0);
    k_pack_b<<<(pb16 + 255) / 256, 256, 0, stream>>>(W_cfg, Wp + 1 * 65536, 16, 0);
    k_pack_b<<<(pb16 + 255) / 256, 256, 0, stream>>>(W_dfg, Wp + 2 * 65536, 16, 0);
    k_pack_b<<<(pb48 + 255) / 256, 256, 0, stream>>>(Wi, WiTp, 48, 1);  // B = Wi^T
    k_pack_b<<<(pb48 + 255) / 256, 256, 0, stream>>>(Wh, WhTp, 48, 1);  // B = Wh^T

    const long NHt = (long)NN * HH;
    const float* hcur = x;
    float* houts[3] = { hA, hB, (float*)d_out };

    for (int L = 0; L < 3; ++L) {
        k_cvt_bf16<<<(int)((NHt + 255) / 256), 256, 0, stream>>>(hcur, hbf, NHt);

        dim3 gg(MTILES, 3, 1);
        k_gemm_xw<<<gg, 256, 0, stream>>>(hbf, Wp, xw);

        k_init_agg<<<NN, 256, 0, stream>>>(xw, dinv, b_ast, b_cfg, b_dfg, agg);

        dim3 gs(EE / 8, 1, 3);
        k_scatter<<<gs, 256, 0, stream>>>(e_ast, e_cfg, e_dfg, dinv, xw, agg);

        k_cvt_bf16<<<(int)((NHt + 255) / 256), 256, 0, stream>>>(agg, abf, NHt);

        k_gru<<<MTILES, 256, 0, stream>>>(abf, hbf, hcur, WiTp, WhTp, bi, bh, houts[L]);
        hcur = houts[L];
    }
}
